// GPRGNNEncoder_82566451298967
// MI455X (gfx1250) — compile-verified, matrix operand order checked by reading, not run on hardware
//
#include <hip/hip_runtime.h>
#include <hip/hip_bf16.h>

// GPR-GNN encoder for MI455X (gfx1250, wave32).
//  Pass A: transpose+convert W1,W2 to bf16 (contiguous per-output-column rows).
//  Pass B: GEMM1 relu(x@W1+b1) -> h1 (bf16), via v_wmma_f32_16x16x32_bf16.
//  Pass C: GEMM2 h1@W2+b2 -> H (fp32) into hop buffer A.
//  Pass D: degrees (self-loop folded), dinv = rsqrt(deg).
//  Pass E: out = gamma0*H; K hops of L2-resident gather/scatter SpMM,
//          out += gamma_k * hop_k.  (100k x 128 x 4B = 51.2MB fits in 192MB L2.)

typedef __attribute__((ext_vector_type(16))) __bf16 v16bf;
typedef __attribute__((ext_vector_type(8)))  __bf16 v8bf;
typedef __attribute__((ext_vector_type(8)))  float  v8f;

#define D_IN 256
#define D_H  512
#define D_E  128

static __device__ __forceinline__ v8bf cvt_f4x2(float4 u, float4 v) {
  v8bf r;
  r[0] = (__bf16)u.x; r[1] = (__bf16)u.y; r[2] = (__bf16)u.z; r[3] = (__bf16)u.w;
  r[4] = (__bf16)v.x; r[5] = (__bf16)v.y; r[6] = (__bf16)v.z; r[7] = (__bf16)v.w;
  return r;
}
static __device__ __forceinline__ v16bf cat8(v8bf lo, v8bf hi) {
  return __builtin_shufflevector(lo, hi, 0,1,2,3,4,5,6,7,8,9,10,11,12,13,14,15);
}

// out[n*K + k] = (bf16) in[k*C + n]   (in: [K x C] fp32, out: [C x K] bf16)
__global__ void __launch_bounds__(256) wtrans_kernel(const float* __restrict__ in,
                                                     __bf16* __restrict__ out,
                                                     int K, int C) {
  int idx = blockIdx.x * 256 + threadIdx.x;
  if (idx < K * C) {
    int n = idx / K;
    int k = idx - n * K;
    out[idx] = (__bf16)in[(size_t)k * C + n];
  }
}

// GEMM1: h1[N, D_H] = relu(x[N, D_IN] @ W1 + b1), bf16 WMMA, one wave per 16x16 tile.
// Grid is exact: waves = (N/16) * (D_H/16); EXEC all-ones at every WMMA.
__global__ void __launch_bounds__(256) gemm1_kernel(const float* __restrict__ x,
                                                    const __bf16* __restrict__ w1t,
                                                    const float* __restrict__ bias1,
                                                    __bf16* __restrict__ h1) {
  const int wave    = (blockIdx.x * 256 + threadIdx.x) >> 5;
  const int lane    = threadIdx.x & 31;
  const int colTile = wave & 31;        // D_H/16 = 32 column tiles
  const int rowTile = wave >> 5;
  const int m    = lane & 15;
  const int half = lane >> 4;

  const float*  xr = x   + (size_t)(rowTile * 16 + m) * D_IN;
  const __bf16* wr = w1t + (size_t)(colTile * 16 + m) * D_IN;

  v8f c = {};
#pragma unroll
  for (int kb = 0; kb < D_IN; kb += 32) {
    const int kA = kb + half * 8;       // lanes 0-15: K {kb..+7, kb+16..+23}; 16-31: +8
    float4 a0 = *(const float4*)(xr + kA);
    float4 a1 = *(const float4*)(xr + kA + 4);
    float4 a2 = *(const float4*)(xr + kA + 16);
    float4 a3 = *(const float4*)(xr + kA + 20);
    v16bf a = cat8(cvt_f4x2(a0, a1), cvt_f4x2(a2, a3));
    v16bf b = cat8(*(const v8bf*)(wr + kA), *(const v8bf*)(wr + kA + 16));
    c = __builtin_amdgcn_wmma_f32_16x16x32_bf16(false, a, false, b, (short)0, c,
                                                false, false);
  }
  const int   nG = colTile * 16 + m;
  const float bv = bias1[nG];
#pragma unroll
  for (int r = 0; r < 8; ++r) {
    const int row = rowTile * 16 + r + 8 * half;   // C/D layout: M = r + 8*half
    float v = c[r] + bv;
    v = v > 0.0f ? v : 0.0f;
    h1[(size_t)row * D_H + nG] = (__bf16)v;
  }
}

// GEMM2: H[N, D_E] = h1[N, D_H] @ W2 + b2, fp32 output.
__global__ void __launch_bounds__(256) gemm2_kernel(const __bf16* __restrict__ h1,
                                                    const __bf16* __restrict__ w2t,
                                                    const float* __restrict__ bias2,
                                                    float* __restrict__ hOut) {
  const int wave    = (blockIdx.x * 256 + threadIdx.x) >> 5;
  const int lane    = threadIdx.x & 31;
  const int colTile = wave & 7;         // D_E/16 = 8 column tiles
  const int rowTile = wave >> 3;
  const int m    = lane & 15;
  const int half = lane >> 4;

  const __bf16* ar = h1  + (size_t)(rowTile * 16 + m) * D_H;
  const __bf16* wr = w2t + (size_t)(colTile * 16 + m) * D_H;

  v8f c = {};
#pragma unroll
  for (int kb = 0; kb < D_H; kb += 32) {
    const int kA = kb + half * 8;
    v16bf a = cat8(*(const v8bf*)(ar + kA), *(const v8bf*)(ar + kA + 16));
    v16bf b = cat8(*(const v8bf*)(wr + kA), *(const v8bf*)(wr + kA + 16));
    c = __builtin_amdgcn_wmma_f32_16x16x32_bf16(false, a, false, b, (short)0, c,
                                                false, false);
  }
  const int   nG = colTile * 16 + m;
  const float bv = bias2[nG];
#pragma unroll
  for (int r = 0; r < 8; ++r) {
    const int row = rowTile * 16 + r + 8 * half;
    hOut[(size_t)row * D_E + nG] = c[r] + bv;
  }
}

// deg[i] = 1 (self-loop folded in)
__global__ void __launch_bounds__(256) deg_init_kernel(float* __restrict__ deg, int n) {
  int i = blockIdx.x * 256 + threadIdx.x;
  if (i < n) deg[i] = 1.0f;
}
__global__ void __launch_bounds__(256) deg_edge_kernel(const int* __restrict__ dst,
                                                       float* __restrict__ deg, int e) {
  int i = blockIdx.x * 256 + threadIdx.x;
  if (i < e) atomicAdd(&deg[dst[i]], 1.0f);
}
__global__ void __launch_bounds__(256) deg_rsqrt_kernel(float* __restrict__ deg, int n) {
  int i = blockIdx.x * 256 + threadIdx.x;
  if (i < n) deg[i] = rsqrtf(deg[i]);  // deg >= 1 always (self-loop)
}

// nxt[i] = dinv[node]^2 * cur[i]  (self-loop contribution, no atomics)
__global__ void __launch_bounds__(256) hop_init_kernel(const float* __restrict__ dinv,
                                                       const float* __restrict__ cur,
                                                       float* __restrict__ nxt, int n128) {
  int i = blockIdx.x * 256 + threadIdx.x;
  if (i < n128) {
    int node = i >> 7;
    float d = dinv[node];
    nxt[i] = d * d * cur[i];
  }
}

// One wave (32 lanes) per edge; each lane handles 4 of the 128 features.
// Feature matrix (51.2MB) is L2-resident on MI455X -> gathers+atomics stay in L2.
__global__ void __launch_bounds__(256) hop_edge_kernel(const int* __restrict__ src,
                                                       const int* __restrict__ dst,
                                                       const float* __restrict__ dinv,
                                                       const float* __restrict__ cur,
                                                       float* __restrict__ nxt, int e) {
  int tid  = blockIdx.x * 256 + threadIdx.x;
  int eidx = tid >> 5;
  if (eidx >= e) return;
  int lane = tid & 31;
  int s = src[eidx];
  int t = dst[eidx];
  float w = dinv[s] * dinv[t];
  const float4 v = *(const float4*)(cur + (size_t)s * D_E + lane * 4);
  float* o = nxt + (size_t)t * D_E + lane * 4;
  atomicAdd(o + 0, w * v.x);
  atomicAdd(o + 1, w * v.y);
  atomicAdd(o + 2, w * v.z);
  atomicAdd(o + 3, w * v.w);
}

// out = (first ? g*h : out + g*h), g = gamma[k] read on device
__global__ void __launch_bounds__(256) accum_kernel(const float* __restrict__ gamma, int k,
                                                    const float* __restrict__ h,
                                                    float* __restrict__ out, int n128,
                                                    int first) {
  int i = blockIdx.x * 256 + threadIdx.x;
  if (i < n128) {
    float g = gamma[k];
    out[i] = first ? g * h[i] : out[i] + g * h[i];
  }
}

static inline size_t alignUp(size_t v) { return (v + 255) & ~(size_t)255; }

extern "C" void kernel_launch(void* const* d_in, const int* in_sizes, int n_in,
                              void* d_out, int out_size, void* d_ws, size_t ws_size,
                              hipStream_t stream) {
  const float* x     = (const float*)d_in[0];
  const int*   ei    = (const int*)d_in[1];
  const float* W1    = (const float*)d_in[2];
  const float* b1    = (const float*)d_in[3];
  const float* W2    = (const float*)d_in[4];
  const float* b2    = (const float*)d_in[5];
  const float* gamma = (const float*)d_in[6];
  float*       out   = (float*)d_out;

  const int N  = in_sizes[0] / D_IN;     // 100000
  const int E  = in_sizes[1] / 2;        // 1600000
  const int Kh = in_sizes[6] - 1;        // 10 hops
  const int* src = ei;
  const int* dst = ei + E;

  // Workspace layout
  char* ws = (char*)d_ws;
  size_t off = 0;
  __bf16* h1b  = (__bf16*)(ws + off); off += alignUp((size_t)N * D_H * sizeof(__bf16));
  __bf16* w1t  = (__bf16*)(ws + off); off += alignUp((size_t)D_H * D_IN * sizeof(__bf16));
  __bf16* w2t  = (__bf16*)(ws + off); off += alignUp((size_t)D_E * D_H * sizeof(__bf16));
  float*  dinv = (float*)(ws + off);  off += alignUp((size_t)N * sizeof(float));
  float*  bufA = (float*)(ws + off);  off += alignUp((size_t)N * D_E * sizeof(float));
  float*  bufB = (float*)(ws + off);  off += alignUp((size_t)N * D_E * sizeof(float));
  if (off > ws_size) return;  // workspace too small: do nothing (deterministic)

  const int n128 = N * D_E;

  // Pass A: weight transpose + bf16 convert
  wtrans_kernel<<<(D_IN * D_H + 255) / 256, 256, 0, stream>>>(W1, w1t, D_IN, D_H);
  wtrans_kernel<<<(D_H * D_E + 255) / 256, 256, 0, stream>>>(W2, w2t, D_H, D_E);

  // Pass B: GEMM1 (exact grid: (N/16)*(D_H/16) waves, 8 waves/block)
  {
    int tiles = (N / 16) * (D_H / 16);
    gemm1_kernel<<<tiles / 8, 256, 0, stream>>>(x, w1t, b1, h1b);
  }
  // Pass C: GEMM2 -> bufA (H)
  {
    int tiles = (N / 16) * (D_E / 16);
    gemm2_kernel<<<tiles / 8, 256, 0, stream>>>(h1b, w2t, b2, bufA);
  }

  // Pass D: degrees -> dinv
  deg_init_kernel<<<(N + 255) / 256, 256, 0, stream>>>(dinv, N);
  deg_edge_kernel<<<(E + 255) / 256, 256, 0, stream>>>(dst, dinv, E);
  deg_rsqrt_kernel<<<(N + 255) / 256, 256, 0, stream>>>(dinv, N);

  // Pass E: out = gamma[0] * H, then K hops
  accum_kernel<<<(n128 + 255) / 256, 256, 0, stream>>>(gamma, 0, bufA, out, n128, 1);

  const int edgeBlocks = (E * 32 + 255) / 256;
  for (int k = 1; k <= Kh; ++k) {
    float* cur = (k & 1) ? bufA : bufB;
    float* nxt = (k & 1) ? bufB : bufA;
    hop_init_kernel<<<(n128 + 255) / 256, 256, 0, stream>>>(dinv, cur, nxt, n128);
    hop_edge_kernel<<<edgeBlocks, 256, 0, stream>>>(src, dst, dinv, cur, nxt, E);
    accum_kernel<<<(n128 + 255) / 256, 256, 0, stream>>>(gamma, k, nxt, out, n128, 0);
  }
}
